// RelativeMultiHeadAttention_1941325218037
// MI455X (gfx1250) — matrix operand order repaired and checked by reading, not compiled, not run on hardware
//
#include <hip/hip_runtime.h>
#include <hip/hip_bf16.h>
#include <stdint.h>

// ---------------------------------------------------------------------------
// Types for CDNA5 WMMA (wave32, 16x16x32 bf16 -> f32)
// ---------------------------------------------------------------------------
typedef __bf16 bf16;
typedef __attribute__((ext_vector_type(16))) __bf16 v16bf;
typedef __attribute__((ext_vector_type(8)))  __bf16 bf16x8;
typedef __attribute__((ext_vector_type(4)))  __bf16 bf16x4;
typedef __attribute__((ext_vector_type(8)))  float  v8f;

#define NHEADS   16
#define HDIM     64
#define EMBED    1024
#define QLEN_    1024
#define MLEN_    1024
#define RLEN_    2048
#define BSZ_     4
#define D3       3072   // 3*EMBED

#define WMMA_BF16(A, B, C) \
  __builtin_amdgcn_wmma_f32_16x16x32_bf16(false, (A), false, (B), (short)0, (C), false, false)

#if defined(__has_builtin) && __has_builtin(__builtin_amdgcn_s_wait_asynccnt)
#define WAIT_ASYNC0() __builtin_amdgcn_s_wait_asynccnt(0)
#else
#define WAIT_ASYNC0() asm volatile("s_wait_asynccnt 0x0" ::: "memory")
#endif

#if defined(__has_builtin) && __has_builtin(__builtin_amdgcn_s_wait_tensorcnt)
#define WAIT_TENSOR0() __builtin_amdgcn_s_wait_tensorcnt(0)
#else
#define WAIT_TENSOR0() asm volatile("s_wait_tensorcnt 0x0" ::: "memory")
#endif

// Async global->LDS 16B copy (GVS addressing: saddr base + 32-bit voffset).
// Tracked by ASYNCcnt; LDS destination address in a VGPR.
#define ASYNC_B128(ldsaddr, voff, sbase)                                  \
  asm volatile("global_load_async_to_lds_b128 %0, %1, %2"                 \
               :: "v"(ldsaddr), "v"(voff), "s"(sbase) : "memory")

// ---------------------------------------------------------------------------
// Tensor Data Mover: 2D tile load global->LDS (128 rows x 32 bf16),
// LDS padding 16B after every 64B stored (=> 40-element row pitch).
// ---------------------------------------------------------------------------
#if defined(__has_builtin)
#if __has_builtin(__builtin_amdgcn_tensor_load_to_lds)
#define HAVE_TDM 1
#endif
#endif

#ifdef HAVE_TDM
typedef __attribute__((ext_vector_type(4))) unsigned int tdm_u32x4;
typedef __attribute__((ext_vector_type(4))) int          tdm_i32x4;
typedef __attribute__((ext_vector_type(8))) int          tdm_i32x8;

static __device__ inline void tdm_load_tile_128x32(uint32_t lds_addr,
                                                   uint64_t gaddr,
                                                   int stride_elems) {
  tdm_u32x4 g0;
  g0[0] = 1u;                                        // count=1, user descriptor
  g0[1] = lds_addr;                                  // LDS byte address
  g0[2] = (uint32_t)(gaddr & 0xFFFFFFFFu);           // global_addr[31:0]
  g0[3] = (uint32_t)((gaddr >> 32) & 0x01FFFFFFu)    // global_addr[56:32]
        | (2u << 30);                                // type = 2
  tdm_i32x8 g1;
  g1[0] = (int)((1u << 16)        // data_size = 2 bytes
              | (1u << 20)        // pad_enable
              | (3u << 22)        // pad_interval: 16 DWORDs (64B)
              | (3u << 25));      // pad_amount:   4 DWORDs (16B)
  g1[1] = (int)(32u << 16);       // tensor_dim0 = 32 (lo half)
  g1[2] = (int)(128u << 16);      // tensor_dim0 hi=0 | tensor_dim1 lo = 128
  g1[3] = (int)(32u << 16);       // tensor_dim1 hi=0 | tile_dim0 = 32
  g1[4] = (int)128;               // tile_dim1 = 128 | tile_dim2 = 0
  g1[5] = stride_elems;           // tensor_dim0_stride lo (data_size units)
  g1[6] = 0;                      // stride hi | tensor_dim1_stride lo
  g1[7] = 0;
  tdm_i32x4 gz;
  gz[0] = 0; gz[1] = 0; gz[2] = 0; gz[3] = 0;
#if __clang_major__ >= 23
  tdm_i32x8 gz8;
  gz8[0] = 0; gz8[1] = 0; gz8[2] = 0; gz8[3] = 0;
  gz8[4] = 0; gz8[5] = 0; gz8[6] = 0; gz8[7] = 0;
  __builtin_amdgcn_tensor_load_to_lds(g0, g1, gz, gz, gz8, 0);
#else
  __builtin_amdgcn_tensor_load_to_lds(g0, g1, gz, gz, 0);
#endif
}
#endif  // HAVE_TDM

// Build a 16-element A/B fragment from two contiguous 8-element (16B) runs.
static __device__ inline v16bf frag16(const bf16* lo, const bf16* hi) {
  bf16x8 a = *(const bf16x8*)lo;
  bf16x8 b = *(const bf16x8*)hi;
  v16bf f;
#pragma unroll
  for (int e = 0; e < 8; ++e) { f[e] = a[e]; f[e + 8] = b[e]; }
  return f;
}
static __device__ inline v16bf fragz() {
  v16bf f;
#pragma unroll
  for (int e = 0; e < 16; ++e) f[e] = (bf16)0.0f;
  return f;
}
static __device__ inline v8f vzero8() {
  v8f z;
#pragma unroll
  for (int e = 0; e < 8; ++e) z[e] = 0.0f;
  return z;
}

// ---------------------------------------------------------------------------
// fp32 -> bf16 cast (vectorized x4, grid-stride)
// ---------------------------------------------------------------------------
__global__ void __launch_bounds__(256)
cast_f32_bf16_x4(const float* __restrict__ src, bf16* __restrict__ dst, long n4) {
  long i = (long)blockIdx.x * blockDim.x + threadIdx.x;
  const long stride = (long)gridDim.x * blockDim.x;
  for (; i < n4; i += stride) {
    const float4 f = ((const float4*)src)[i];
    bf16x4 o;
    o[0] = (bf16)f.x; o[1] = (bf16)f.y; o[2] = (bf16)f.z; o[3] = (bf16)f.w;
    ((bf16x4*)dst)[i] = o;
  }
}

// ---------------------------------------------------------------------------
// Tiled bf16 WMMA GEMM: C[M,N] = A[M,K] @ B[K,N], A/B bf16 row-major.
// Block tile 128x128, K-step 32, 8 waves (4x2), each wave 32x64.
// Double-buffered LDS. A tile: TDM descriptor load (TENSORcnt), one issue per
// block (wave 0); fallback: async global->LDS. B tile: async global->LDS into
// raw buffer (ASYNCcnt), then LDS->LDS transpose after the WMMA block.
// OUTF32: 1 -> fp32 output, 0 -> bf16 output.
// ---------------------------------------------------------------------------
template <int OUTF32>
__global__ void __launch_bounds__(256)
gemm_bf16_wmma(const bf16* __restrict__ A, const bf16* __restrict__ B,
               void* __restrict__ Cv, int M, int N, int K) {
  __shared__ bf16 Al[2][128][40];    // [buf][row][k], pitch-padded (80B)
  __shared__ bf16 Bl[2][128][40];    // [buf][col][k] (transposed), padded
  __shared__ bf16 Braw[2][32][128];  // [buf][k][col] raw async landing zone

  const int tid  = threadIdx.x;
  const int wave = tid >> 5, lane = tid & 31;
  const int wm = wave & 3, wn = wave >> 2;     // 4 (M) x 2 (N) wave grid
  const int lm = lane & 15, hh = lane >> 4;    // half-wave select
  const int kb8 = hh * 8, kb16 = hh * 16;
  const int i0 = blockIdx.y * 128, j0 = blockIdx.x * 128;

  v8f acc[2][4];
#pragma unroll
  for (int a = 0; a < 2; ++a)
#pragma unroll
    for (int f = 0; f < 4; ++f) acc[a][f] = vzero8();

  const int ar = tid >> 1, ah = (tid & 1) * 16;   // A tile: 16 elems/thread
  const int bk = tid >> 3, bn = (tid & 7) * 16;   // B tile: 16 elems/thread

  // Loop-invariant async addressing.
  const uint32_t bvoff = (uint32_t)((bk * N + bn) * (int)sizeof(bf16));
  uint32_t ldsBraw[2];
  ldsBraw[0] = (uint32_t)(uintptr_t)&Braw[0][bk][bn];
  ldsBraw[1] = (uint32_t)(uintptr_t)&Braw[1][bk][bn];
#ifndef HAVE_TDM
  const uint32_t avoff = (uint32_t)((ar * K + ah) * (int)sizeof(bf16));
  uint32_t ldsA[2];
  ldsA[0] = (uint32_t)(uintptr_t)&Al[0][ar][ah];
  ldsA[1] = (uint32_t)(uintptr_t)&Al[1][ar][ah];
#endif

  auto stageA = [&](int bufi, int kk) {
#ifdef HAVE_TDM
    if (wave == 0) {
      tdm_load_tile_128x32((uint32_t)(uintptr_t)&Al[bufi][0][0],
                           (uint64_t)(uintptr_t)(A + (long)i0 * K + kk), K);
    }
#else
    const uint64_t gbase = (uint64_t)(uintptr_t)(A + (long)i0 * K + kk);
    ASYNC_B128(ldsA[bufi],       avoff,       gbase);
    ASYNC_B128(ldsA[bufi] + 16u, avoff + 16u, gbase);
#endif
  };
  auto stageB = [&](int bufi, int kk) {
    const uint64_t gbase = (uint64_t)(uintptr_t)(B + (long)kk * N + j0);
    ASYNC_B128(ldsBraw[bufi],       bvoff,       gbase);
    ASYNC_B128(ldsBraw[bufi] + 16u, bvoff + 16u, gbase);
  };
  // Each thread transposes exactly the 32B it async-copied itself, so
  // s_wait_asynccnt on its own copies is sufficient before these reads.
  auto finishB = [&](int bufi) {
    bf16x8 t0 = *(const bf16x8*)&Braw[bufi][bk][bn];
    bf16x8 t1 = *(const bf16x8*)&Braw[bufi][bk][bn + 8];
#pragma unroll
    for (int e = 0; e < 8; ++e) {
      Bl[bufi][bn + e][bk]     = t0[e];
      Bl[bufi][bn + 8 + e][bk] = t1[e];
    }
  };

  // ---- prologue: stage tile 0 ----
  stageA(0, 0);
  stageB(0, 0);
  WAIT_ASYNC0();
  finishB(0);
  WAIT_TENSOR0();
  __syncthreads();

  for (int k0 = 0; k0 < K; k0 += 32) {
    const int buf = (k0 >> 5) & 1;
    const bool has_next = (k0 + 32 < K);

    // ---- issue next-tile staging (overlaps with WMMA below) ----
    if (has_next) {
      stageA(buf ^ 1, k0 + 32);
      stageB(buf ^ 1, k0 + 32);
    }

    // ---- fragments + WMMA on current buffer ----
    v16bf af[2], bfv[4];
#pragma unroll
    for (int a = 0; a < 2; ++a) {
      const bf16* p = &Al[buf][wm * 32 + a * 16 + lm][0];
      af[a] = frag16(p + kb16, p + kb16 + 8);       // contiguous-K A layout
    }
#pragma unroll
    for (int f = 0; f < 4; ++f) {
      const bf16* p = &Bl[buf][wn * 64 + f * 16 + lm][kb16];
      bfv[f] = frag16(p, p + 8);                    // contiguous-K B layout
    }
#pragma unroll
    for (int a = 0; a < 2; ++a)
#pragma unroll
      for (int f = 0; f < 4; ++f)
        acc[a][f] = WMMA_BF16(af[a], bfv[f], acc[a][f]);

    // ---- finish next-tile B transpose; wait own DMAs; barrier ----
    WAIT_ASYNC0();
    if (has_next) finishB(buf ^ 1);
    WAIT_TENSOR0();
    __syncthreads();
  }

  // ---- epilogue: C layout (lane = col, VGPR g = row g / g+8) ----
#pragma unroll
  for (int a = 0; a < 2; ++a)
#pragma unroll
    for (int f = 0; f < 4; ++f)
#pragma unroll
      for (int g = 0; g < 8; ++g) {
        const int row = i0 + wm * 32 + a * 16 + g + hh * 8;
        const int col = j0 + wn * 64 + f * 16 + lm;
        if (OUTF32) ((float*)Cv)[(long)row * N + col] = acc[a][f][g];
        else        ((bf16*)Cv)[(long)row * N + col] = (bf16)acc[a][f][g];
      }
}

// ---------------------------------------------------------------------------
// Flash-style relative attention. One wave per (b, head, 16-query tile).
// wheads: [klen*BSZ, 3072] bf16 rows = t*BSZ+b, cols = [q | k | v] per head.
// rk:     [RLEN, 1024] bf16.   avec out: [(q*BSZ+b), 1024] bf16.
// ---------------------------------------------------------------------------
__global__ void __launch_bounds__(128)
attn_relshift_wmma(const bf16* __restrict__ wheads, const bf16* __restrict__ rk,
                   const float* __restrict__ u, const float* __restrict__ vv,
                   bf16* __restrict__ avec) {
  __shared__ float Sac[4][16][32];
  __shared__ float Tb[4][16][48];
  __shared__ bf16  Pl[4][16][32];
  __shared__ bf16  Vl[4][64][40];
  __shared__ float mrow[4][16], lrow[4][16], arow[4][16];

  const int tid = threadIdx.x;
  const int w = tid >> 5, lane = tid & 31;
  const int wid = blockIdx.x * 4 + w;
  const int b  = wid & 3;
  const int nh = (wid >> 2) & 15;
  const int i0 = (wid >> 6) * 16;
  const int lm = lane & 15, hh = lane >> 4;
  const int kb8 = hh * 8, kb16 = hh * 16;
  const float scale = 0.125f;  // 1/sqrt(64)

  if (lane < 16) { mrow[w][lane] = -1e30f; lrow[w][lane] = 0.0f; }

  // ---- build Q fragments: q + u (AC) and q + v (BD), A-matrix layout ----
  v16bf qac[2], qbd[2];
  {
    const int  qi   = i0 + lm;
    const long qrow = (long)((MLEN_ + qi) * BSZ_ + b) * D3 + nh * HDIM;
#pragma unroll
    for (int s = 0; s < 2; ++s) {
      v16bf fa, fb;
#pragma unroll
      for (int e = 0; e < 16; ++e) {
        const int d = s * 32 + kb16 + e;
        const float wq = (float)wheads[qrow + d];
        fa[e] = (bf16)(wq + u[nh * HDIM + d]);
        fb[e] = (bf16)(wq + vv[nh * HDIM + d]);
      }
      qac[s] = fa; qbd[s] = fb;
    }
  }

  v8f o[4];
#pragma unroll
  for (int f = 0; f < 4; ++f) o[f] = vzero8();

  for (int j0 = 0; j0 < RLEN_; j0 += 32) {
    // prefetch next block's K/V rows into cache (global_prefetch_b8)
    if (j0 + 32 < RLEN_) {
      const long prow = (long)((j0 + 32 + lane) * BSZ_ + b) * D3;
      __builtin_prefetch(wheads + prow + EMBED + nh * HDIM, 0, 0);
      __builtin_prefetch(wheads + prow + 2 * EMBED + nh * HDIM, 0, 0);
    }
    // ---- stage V block transposed into LDS: Vl[d][key] ----
    {
      const long vrow = (long)((j0 + lane) * BSZ_ + b) * D3 + 2 * EMBED + nh * HDIM;
#pragma unroll
      for (int d8 = 0; d8 < 8; ++d8) {
        bf16x8 chunk = *(const bf16x8*)(wheads + vrow + d8 * 8);
#pragma unroll
        for (int e = 0; e < 8; ++e) Vl[w][d8 * 8 + e][lane] = chunk[e];
      }
    }
    // ---- AC scores: S = Qac @ K^T (16x32) ----
#pragma unroll
    for (int h = 0; h < 2; ++h) {
      v8f acc = vzero8();
      const int  j    = j0 + h * 16 + lm;
      const long krow = (long)(j * BSZ_ + b) * D3 + EMBED + nh * HDIM;
#pragma unroll
      for (int s = 0; s < 2; ++s) {
        v16bf bfv = frag16(wheads + krow + s * 32 + kb16,
                           wheads + krow + s * 32 + kb16 + 8);
        acc = WMMA_BF16(qac[s], bfv, acc);
      }
#pragma unroll
      for (int g = 0; g < 8; ++g) Sac[w][g + hh * 8][h * 16 + lm] = acc[g];
    }
    // ---- BD band: T[i][u] = Qbd[i] . rk[base+u], u in [0,48) ----
    const int base = j0 + (QLEN_ - 16) - i0;
#pragma unroll
    for (int c = 0; c < 3; ++c) {
      v8f acc = vzero8();
      const int  t    = base + c * 16 + lm;
      const bool ok   = (t < RLEN_);
      const long rrow = (long)t * EMBED + nh * HDIM;
#pragma unroll
      for (int s = 0; s < 2; ++s) {
        v16bf bfv = ok ? frag16(rk + rrow + s * 32 + kb16,
                                rk + rrow + s * 32 + kb16 + 8)
                       : fragz();
        acc = WMMA_BF16(qbd[s], bfv, acc);
      }
#pragma unroll
      for (int g = 0; g < 8; ++g) Tb[w][g + hh * 8][c * 16 + lm] = acc[g];
    }
    asm volatile("s_wait_dscnt 0x0" ::: "memory");  // in-wave LDS fence

    // ---- online softmax: lane r handles query row r; rel-shift select ----
    if (lane < 16) {
      const int r = lane;
      const float mold = mrow[w][r], lold = lrow[w][r];
      float mx = mold;
      float sv[32];
#pragma unroll
      for (int jj = 0; jj < 32; ++jj) {
        const float s = (Sac[w][r][jj] + Tb[w][r][15 - r + jj]) * scale;
        sv[jj] = s;
        mx = fmaxf(mx, s);
      }
      const float al = __expf(mold - mx);
      float ls = lold * al;
#pragma unroll
      for (int jj = 0; jj < 32; ++jj) {
        const float p = __expf(sv[jj] - mx);
        ls += p;
        Pl[w][r][jj] = (bf16)p;
      }
      mrow[w][r] = mx; lrow[w][r] = ls; arow[w][r] = al;
    }
    asm volatile("s_wait_dscnt 0x0" ::: "memory");

    // ---- rescale O, then O += P @ V ----
    float al8[8];
#pragma unroll
    for (int g = 0; g < 8; ++g) al8[g] = arow[w][g + hh * 8];
#pragma unroll
    for (int f = 0; f < 4; ++f)
#pragma unroll
      for (int g = 0; g < 8; ++g) o[f][g] *= al8[g];

    v16bf pf;
    {
      const bf16* p = &Pl[w][lm][0];
      pf = frag16(p + kb16, p + kb16 + 8);
    }
#pragma unroll
    for (int f = 0; f < 4; ++f) {
      const bf16* p = &Vl[w][f * 16 + lm][kb16];
      v16bf vf = frag16(p, p + 8);
      o[f] = WMMA_BF16(pf, vf, o[f]);
    }
  }

  // ---- epilogue: normalize and store attn_vec (bf16) ----
  float li8[8];
#pragma unroll
  for (int g = 0; g < 8; ++g) li8[g] = 1.0f / lrow[w][g + hh * 8];
#pragma unroll
  for (int f = 0; f < 4; ++f)
#pragma unroll
    for (int g = 0; g < 8; ++g) {
      const int qi = i0 + g + hh * 8;
      const int d  = f * 16 + lm;
      avec[(long)(qi * BSZ_ + b) * EMBED + nh * HDIM + d] = (bf16)(o[f][g] * li8[g]);
    }
}

// ---------------------------------------------------------------------------
// Residual + LayerNorm: out = LN(x + attn_out) * gamma + beta
// ---------------------------------------------------------------------------
__global__ void __launch_bounds__(256)
residual_ln(const float* __restrict__ x, const float* __restrict__ aout,
            const float* __restrict__ gamma, const float* __restrict__ beta,
            float* __restrict__ out) {
  const int row = blockIdx.x;
  const int t = threadIdx.x;
  __shared__ float ssum[256], ssq[256];
  const long base = (long)row * EMBED;
  float h[4], sum = 0.0f, sq = 0.0f;
#pragma unroll
  for (int k = 0; k < 4; ++k) {
    const int d = t + k * 256;
    const float hv = x[base + d] + aout[base + d];
    h[k] = hv; sum += hv; sq += hv * hv;
  }
  ssum[t] = sum; ssq[t] = sq;
  __syncthreads();
  for (int s = 128; s > 0; s >>= 1) {
    if (t < s) { ssum[t] += ssum[t + s]; ssq[t] += ssq[t + s]; }
    __syncthreads();
  }
  const float mean = ssum[0] * (1.0f / EMBED);
  const float var  = ssq[0] * (1.0f / EMBED) - mean * mean;
  const float inv  = rsqrtf(var + 1e-5f);
#pragma unroll
  for (int k = 0; k < 4; ++k) {
    const int d = t + k * 256;
    out[base + d] = (h[k] - mean) * inv * gamma[d] + beta[d];
  }
}

// ---------------------------------------------------------------------------
// Host launcher
// ---------------------------------------------------------------------------
extern "C" void kernel_launch(void* const* d_in, const int* in_sizes, int n_in,
                              void* d_out, int out_size, void* d_ws, size_t ws_size,
                              hipStream_t stream) {
  const float* x    = (const float*)d_in[0];  // [1024,4,1024]
  const float* r    = (const float*)d_in[1];  // [2048,1024]
  const float* u    = (const float*)d_in[2];  // [16,64]
  const float* v    = (const float*)d_in[3];  // [16,64]
  const float* mems = (const float*)d_in[4];  // [1024,4,1024]
  const float* Wqkv = (const float*)d_in[5];  // [1024,3072]
  const float* Wr   = (const float*)d_in[6];  // [1024,1024]
  const float* Wo   = (const float*)d_in[7];  // [1024,1024]
  const float* lgam = (const float*)d_in[8];
  const float* lbet = (const float*)d_in[9];
  float* out = (float*)d_out;

  // Workspace layout (all regions 16B aligned; ~106 MB total)
  bf16* ctx    = (bf16*)d_ws;                 // [8192,1024]  context bf16
  bf16* wqkvb  = ctx    + 8192L * 1024;       // [1024,3072]
  bf16* wrb    = wqkvb  + 1024L * 3072;       // [1024,1024]
  bf16* rb     = wrb    + 1024L * 1024;       // [2048,1024]
  bf16* wob    = rb     + 2048L * 1024;       // [1024,1024]
  bf16* wheads = wob    + 1024L * 1024;       // [8192,3072]  qkv projection
  bf16* rkb    = wheads + 8192L * 3072;       // [2048,1024]  r projection
  bf16* avec   = rkb    + 2048L * 1024;       // [4096,1024]  attn_vec
  float* aout  = (float*)(avec + 4096L * 1024);  // [4096,1024] fp32

  // 1) casts (context = concat(mems, x): layouts are flat-compatible)
  cast_f32_bf16_x4<<<2048, 256, 0, stream>>>(mems, ctx,            4194304 / 4);
  cast_f32_bf16_x4<<<2048, 256, 0, stream>>>(x,    ctx + 4194304L, 4194304 / 4);
  cast_f32_bf16_x4<<<2048, 256, 0, stream>>>(Wqkv, wqkvb,          3145728 / 4);
  cast_f32_bf16_x4<<<1024, 256, 0, stream>>>(Wr,   wrb,            1048576 / 4);
  cast_f32_bf16_x4<<<1024, 256, 0, stream>>>(r,    rb,             2097152 / 4);
  cast_f32_bf16_x4<<<1024, 256, 0, stream>>>(Wo,   wob,            1048576 / 4);

  // 2) qkv projection: [8192,1024] @ [1024,3072] -> bf16
  gemm_bf16_wmma<0><<<dim3(24, 64), 256, 0, stream>>>(ctx, wqkvb, wheads, 8192, 3072, 1024);
  // 3) r projection: [2048,1024] @ [1024,1024] -> bf16
  gemm_bf16_wmma<0><<<dim3(8, 16), 256, 0, stream>>>(rb, wrb, rkb, 2048, 1024, 1024);
  // 4) attention (AC + rel-shifted BD, flash softmax, P@V)
  attn_relshift_wmma<<<1024, 128, 0, stream>>>(wheads, rkb, u, v, avec);
  // 5) output projection: [4096,1024] @ [1024,1024] -> fp32
  gemm_bf16_wmma<1><<<dim3(8, 32), 256, 0, stream>>>(avec, wob, aout, 4096, 1024, 1024);
  // 6) residual + layernorm
  residual_ln<<<4096, 256, 0, stream>>>(x, aout, lgam, lbet, out);
}
